// Mamba2SSD_13271448944884
// MI455X (gfx1250) — matrix operand order repaired and checked by reading, compiled, not run on hardware
//
#include <hip/hip_runtime.h>

// ---------------- problem constants ----------------
#define B_SZ  2
#define L_SZ  4096
#define DM    1024      // d_model
#define DI    2048      // d_inner
#define DS    64        // d_state
#define NCH   32        // scan chunks
#define CLEN  128       // chunk length (NCH*CLEN == L_SZ)
#define BLR   (B_SZ*L_SZ)   // 8192 GEMM rows

typedef __attribute__((ext_vector_type(8)))  float  v8f;
typedef __attribute__((ext_vector_type(8)))  __bf16 v8bf;
typedef __attribute__((ext_vector_type(16))) __bf16 v16bf;
typedef unsigned int u32x4 __attribute__((ext_vector_type(4)));
typedef int          i32x4 __attribute__((ext_vector_type(4)));
typedef int          i32x8 __attribute__((ext_vector_type(8)));

__device__ __forceinline__ unsigned short f32_to_bf16_bits(float f) {
  union { float f; unsigned u; } v; v.f = f;
  unsigned r = v.u + 0x7FFFu + ((v.u >> 16) & 1u);   // round-to-nearest-even
  return (unsigned short)(r >> 16);
}

// ---------------- TDM: async 2-D bf16 tile load, global -> LDS ----------------
// Builds the Tensor DMA Descriptor (D#) per CDNA5 ISA ch.8:
//  group0: [1:0]=count=1, [63:32]=lds_addr, [120:64]=global_addr, [127:126]=type=2
//  group1: [17:16]=data_size(1 => 2B), [79:48]=tensor_dim0, [111:80]=tensor_dim1,
//          [127:112]=tile_dim0, [143:128]=tile_dim1, [207:160]=tensor_dim0_stride
//  groups 2/3: zero (2-D tensor). Tracked with TENSORcnt.
__device__ __forceinline__ void tdm_load_tile_bf16(
    const unsigned short* gsrc, unsigned int lds_byte_addr,
    unsigned int tile_w_elems, unsigned int tile_h_rows,
    unsigned int row_stride_elems) {
  const unsigned long long ga = (unsigned long long)(uintptr_t)gsrc;
  u32x4 g0;
  g0[0] = 1u;                                     // count=1, user mode, no gather
  g0[1] = lds_byte_addr;                          // lds_addr
  g0[2] = (unsigned int)(ga & 0xFFFFFFFFu);       // global_addr[31:0]
  g0[3] = (unsigned int)((ga >> 32) & 0x01FFFFFFu) | (2u << 30);  // ga[56:32] | type=2

  const unsigned int td0 = row_stride_elems;      // tensor_dim0 >= tile_w
  const unsigned int td1 = tile_h_rows;           // tensor_dim1
  i32x8 g1;
  g1[0] = (int)(1u << 16);                                        // data_size=1 (2 bytes)
  g1[1] = (int)((td0 & 0xFFFFu) << 16);                           // tensor_dim0 lo16
  g1[2] = (int)(((td0 >> 16) & 0xFFFFu) | ((td1 & 0xFFFFu) << 16));
  g1[3] = (int)(((td1 >> 16) & 0xFFFFu) | (tile_w_elems << 16));  // tile_dim0
  g1[4] = (int)(tile_h_rows & 0xFFFFu);                           // tile_dim1 (tile_dim2=0)
  g1[5] = (int)row_stride_elems;                                  // tensor_dim0_stride lo32
  g1[6] = 0;                                                      // stride hi16 | dim1_stride lo16
  g1[7] = 0;

  i32x4 gz = {0, 0, 0, 0};
#if defined(__clang_major__) && (__clang_major__ >= 23)
  i32x8 gz8 = {0, 0, 0, 0, 0, 0, 0, 0};
  __builtin_amdgcn_tensor_load_to_lds(g0, g1, gz, gz, gz8, 0);
#else
  __builtin_amdgcn_tensor_load_to_lds(g0, g1, gz, gz, 0);
#endif
}

__device__ __forceinline__ unsigned int lds_addr_of(const void* p) {
  return (unsigned int)(uintptr_t)p;   // low 32 bits of generic ptr = LDS byte offset
}

// ---------------- elementwise kernels (grid-stride) ----------------
__global__ void k_cvt_bf16(const float* __restrict__ in, unsigned short* __restrict__ out, long n) {
  long i = (long)blockIdx.x * blockDim.x + threadIdx.x;
  long s = (long)gridDim.x * blockDim.x;
  for (; i < n; i += s) out[i] = f32_to_bf16_bits(in[i]);
}

// x_inner = xg[:, :DI] * silu(xg[:, DI:])
__global__ void k_gate(const float* __restrict__ xg, float* __restrict__ xi, long n) {
  long i = (long)blockIdx.x * blockDim.x + threadIdx.x;
  long s = (long)gridDim.x * blockDim.x;
  for (; i < n; i += s) {
    long r = i / DI; int c = (int)(i % DI);
    float a = xg[r * (2 * DI) + c];
    float g = xg[r * (2 * DI) + DI + c];
    xi[i] = a * (g / (1.0f + __expf(-g)));
  }
}

// dt = softplus(dt_raw + dt_b), in place
__global__ void k_softplus(float* __restrict__ dt, const float* __restrict__ bias, long n) {
  long i = (long)blockIdx.x * blockDim.x + threadIdx.x;
  long s = (long)gridDim.x * blockDim.x;
  for (; i < n; i += s) {
    int c = (int)(i % DI);
    float v = dt[i] + bias[c];
    dt[i] = (v > 20.0f) ? v : log1pf(__expf(v));
  }
}

// y_bf16 = bf16(y + D[d] * x_conv)
__global__ void k_skip_cvt(const float* __restrict__ y, const float* __restrict__ xc,
                           const float* __restrict__ Dw, unsigned short* __restrict__ yb, long n) {
  long i = (long)blockIdx.x * blockDim.x + threadIdx.x;
  long s = (long)gridDim.x * blockDim.x;
  for (; i < n; i += s) {
    int d = (int)(i % DI);
    yb[i] = f32_to_bf16_bits(y[i] + Dw[d] * xc[i]);
  }
}

// ---------------- causal depthwise conv (k=4) + RMSNorm ----------------
__global__ __launch_bounds__(256) void k_conv_rms(
    const float* __restrict__ xi, const float* __restrict__ cw, const float* __restrict__ cb,
    const float* __restrict__ nw, float* __restrict__ xc, unsigned short* __restrict__ xcb) {
  const int bl = blockIdx.x;
  const int b = bl / L_SZ, l = bl % L_SZ;
  float vals[8];
  float ssq = 0.0f;
#pragma unroll
  for (int i = 0; i < 8; i++) {
    const int d = threadIdx.x + i * 256;
    float acc = cb[d];
#pragma unroll
    for (int j = 0; j < 4; j++) {
      const int t = l - 3 + j;
      if (t >= 0) acc += cw[d * 4 + j] * xi[((size_t)b * L_SZ + t) * DI + d];
    }
    vals[i] = acc;
    ssq += acc * acc;
  }
  __shared__ float wsum[8];
  for (int o = 16; o > 0; o >>= 1) ssq += __shfl_xor(ssq, o, 32);
  if ((threadIdx.x & 31) == 0) wsum[threadIdx.x >> 5] = ssq;
  __syncthreads();
  if (threadIdx.x == 0) {
    float t = 0.0f;
    for (int i = 0; i < 8; i++) t += wsum[i];
    wsum[0] = t;
  }
  __syncthreads();
  const float r = rsqrtf(wsum[0] * (1.0f / DI) + 1.1920928955078125e-07f);
  const size_t base = (size_t)bl * DI;
#pragma unroll
  for (int i = 0; i < 8; i++) {
    const int d = threadIdx.x + i * 256;
    const float o = vals[i] * r * nw[d];
    xc[base + d] = o;
    xcb[base + d] = f32_to_bf16_bits(o);
  }
}

// ---------------- WMMA bf16 GEMM with TDM double-buffered LDS ----------------
// C[M,N] = A[M,K] * W[N,K]^T. Block tile 128x128, 8 waves (2x4), wave tile 64x32.
// A/B K-tiles (128x32 bf16 = 8KB each) are streamed by the Tensor Data Mover into
// double-buffered LDS; wave 0 issues the descriptors and gates on TENSORcnt.
#define BM 128
#define BN 128
#define BK 32

__global__ __launch_bounds__(256) void k_gemm_bf16(
    const unsigned short* __restrict__ A,   // [M,K] bf16 bits, row-major
    const unsigned short* __restrict__ W,   // [N,K] bf16 bits, row-major
    float* __restrict__ C,                  // [M,N] fp32
    int M, int N, int K) {
  __shared__ __align__(16) unsigned short As[2][BM * BK];
  __shared__ __align__(16) unsigned short Bs[2][BN * BK];
  const int bm = blockIdx.y * BM;
  const int bn = blockIdx.x * BN;
  const int tid = threadIdx.x;
  const int lane = tid & 31;
  const int wave = tid >> 5;
  const int wm = (wave >> 2) * 64;
  const int wn = (wave & 3) * 32;
  const int ml = lane & 15;
  const int kh = (lane >> 4) * 8;    // A frag k-half offset (ISA 16-bit A layout)
  const int kg = (lane >> 4) * 16;   // unused; B frags use kh, same k order as A

  v8f acc[4][2] = {};

  const bool issuer = (tid == 0);
  const unsigned short* gA = A + (size_t)bm * K;
  const unsigned short* gW = W + (size_t)bn * K;

  // prologue: stage 0
  if (issuer) {
    tdm_load_tile_bf16(gA, lds_addr_of(&As[0][0]), BK, BM, (unsigned)K);
    tdm_load_tile_bf16(gW, lds_addr_of(&Bs[0][0]), BK, BN, (unsigned)K);
    __builtin_amdgcn_s_wait_tensorcnt(0);
  }
  __syncthreads();

  const int niter = K / BK;
  for (int it = 0; it < niter; ++it) {
    const int cur = it & 1;
    const bool more = (it + 1) < niter;
    if (issuer && more) {   // kick off next K-tile while computing on current
      const int kk2 = (it + 1) * BK;
      tdm_load_tile_bf16(gA + kk2, lds_addr_of(&As[cur ^ 1][0]), BK, BM, (unsigned)K);
      tdm_load_tile_bf16(gW + kk2, lds_addr_of(&Bs[cur ^ 1][0]), BK, BN, (unsigned)K);
    }

    v16bf af[4], bfr[2];
#pragma unroll
    for (int tm = 0; tm < 4; tm++) {
      const unsigned short* r = &As[cur][0] + (wm + tm * 16 + ml) * BK;
      v8bf lo = *(const v8bf*)(r + kh);
      v8bf hi = *(const v8bf*)(r + 16 + kh);
      af[tm] = __builtin_shufflevector(lo, hi, 0, 1, 2, 3, 4, 5, 6, 7, 8, 9, 10, 11, 12, 13, 14, 15);
    }
#pragma unroll
    for (int tn = 0; tn < 2; tn++) {
      const unsigned short* r = &Bs[cur][0] + (wn + tn * 16 + ml) * BK + kh;
      v8bf lo = *(const v8bf*)(r);
      v8bf hi = *(const v8bf*)(r + 16);
      bfr[tn] = __builtin_shufflevector(lo, hi, 0, 1, 2, 3, 4, 5, 6, 7, 8, 9, 10, 11, 12, 13, 14, 15);
    }
#pragma unroll
    for (int tm = 0; tm < 4; tm++)
#pragma unroll
      for (int tn = 0; tn < 2; tn++)
        acc[tm][tn] = __builtin_amdgcn_wmma_f32_16x16x32_bf16(
            false, af[tm], false, bfr[tn], (short)0, acc[tm][tn], false, false);

    if (issuer && more) __builtin_amdgcn_s_wait_tensorcnt(0);
    __syncthreads();
  }

  const int rowh = (lane >> 4) * 8;
#pragma unroll
  for (int tm = 0; tm < 4; tm++)
#pragma unroll
    for (int tn = 0; tn < 2; tn++)
#pragma unroll
      for (int r = 0; r < 8; r++) {
        const int row = bm + wm + tm * 16 + rowh + r;
        const int col = bn + wn + tn * 16 + ml;
        C[(size_t)row * N + col] = acc[tm][tn][r];
      }
}

// ---------------- chunked selective scan ----------------
__global__ __launch_bounds__(64) void k_scan_local(
    const float* __restrict__ dt, const float* __restrict__ xc, const float* __restrict__ bct,
    const float* __restrict__ A_log, float* __restrict__ hloc, float* __restrict__ Ploc) {
  const int n = threadIdx.x;     // 0..63
  const int d = blockIdx.x, c = blockIdx.y, b = blockIdx.z;
  const float Ac = -__expf(A_log[d * DS + n]);
  float h = 0.0f, P = 1.0f;
  const int t0 = c * CLEN;
  for (int t = 0; t < CLEN; t++) {
    const size_t row = (size_t)b * L_SZ + t0 + t;
    const float dtv = dt[row * DI + d];
    const float xv  = xc[row * DI + d];
    const float Bv  = bct[row * 128 + n];
    const float a = __expf(dtv * Ac);
    h = a * h + dtv * xv * Bv;
    P *= a;
  }
  const size_t o = (((size_t)b * DI + d) * NCH + c) * DS + n;
  hloc[o] = h;
  Ploc[o] = P;
}

__global__ void k_scan_combine(const float* __restrict__ hloc, const float* __restrict__ Ploc,
                               float* __restrict__ Hinit) {
  const long i = (long)blockIdx.x * blockDim.x + threadIdx.x;
  if (i >= (long)B_SZ * DI * DS) return;
  const int n = (int)(i % DS);
  const long bd = i / DS;
  float H = 0.0f;
  for (int c = 0; c < NCH; c++) {
    const size_t o = ((size_t)bd * NCH + c) * DS + n;
    Hinit[o] = H;
    H = hloc[o] + Ploc[o] * H;
  }
}

__global__ __launch_bounds__(64) void k_scan_emit(
    const float* __restrict__ dt, const float* __restrict__ xc, const float* __restrict__ bct,
    const float* __restrict__ A_log, const float* __restrict__ Hinit, float* __restrict__ y) {
  const int n = threadIdx.x;
  const int d = blockIdx.x, c = blockIdx.y, b = blockIdx.z;
  const float Ac = -__expf(A_log[d * DS + n]);
  float h = Hinit[(((size_t)b * DI + d) * NCH + c) * DS + n];
  __shared__ float red[2];
  const int t0 = c * CLEN;
  for (int t = 0; t < CLEN; t++) {
    const size_t row = (size_t)b * L_SZ + t0 + t;
    const float dtv = dt[row * DI + d];
    const float xv  = xc[row * DI + d];
    const float Bv  = bct[row * 128 + n];
    const float Cv  = bct[row * 128 + 64 + n];
    const float a = __expf(dtv * Ac);
    h = a * h + dtv * xv * Bv;
    float p = h * Cv;
    for (int o2 = 16; o2 > 0; o2 >>= 1) p += __shfl_xor(p, o2, 32);
    if ((n & 31) == 0) red[n >> 5] = p;
    __syncthreads();
    if (n == 0) y[row * DI + d] = red[0] + red[1];
    __syncthreads();
  }
}

// ---------------- launcher ----------------
extern "C" void kernel_launch(void* const* d_in, const int* in_sizes, int n_in,
                              void* d_out, int out_size, void* d_ws, size_t ws_size,
                              hipStream_t stream) {
  const float* x      = (const float*)d_in[0];
  const float* in_w   = (const float*)d_in[1];   // [2*DI, DM]
  const float* conv_w = (const float*)d_in[2];   // [DI,1,4]
  const float* conv_b = (const float*)d_in[3];
  const float* norm_w = (const float*)d_in[4];
  const float* dt_w   = (const float*)d_in[5];   // [DI, DI]
  const float* dt_b   = (const float*)d_in[6];
  const float* A_log  = (const float*)d_in[7];   // [DI, DS]
  const float* Dp     = (const float*)d_in[8];
  const float* B_w    = (const float*)d_in[9];   // [DS, DI]
  const float* C_w    = (const float*)d_in[10];  // [DS, DI]
  const float* out_w  = (const float*)d_in[11];  // [DM, DI]
  float* out = (float*)d_out;

  char* ws = (char*)d_ws;
  size_t off = 0;
  auto alloc = [&](size_t bytes) -> void* {
    off = (off + 255) & ~(size_t)255;
    void* p = ws + off;
    off += bytes;
    return p;
  };

  unsigned short* x_bf    = (unsigned short*)alloc((size_t)BLR * DM * 2);
  unsigned short* w_in_bf = (unsigned short*)alloc((size_t)2 * DI * DM * 2);
  unsigned short* dt_w_bf = (unsigned short*)alloc((size_t)DI * DI * 2);
  unsigned short* bc_w_bf = (unsigned short*)alloc((size_t)128 * DI * 2);
  unsigned short* out_w_bf= (unsigned short*)alloc((size_t)DM * DI * 2);
  float* xg      = (float*)alloc((size_t)BLR * 2 * DI * 4);   // reused: dt | y
  float* x_inner = (float*)alloc((size_t)BLR * DI * 4);       // reused: hloc | Ploc
  float* xc_f32  = (float*)alloc((size_t)BLR * DI * 4);
  unsigned short* xc_bf = (unsigned short*)alloc((size_t)BLR * DI * 2); // reused: y_bf
  float* bct   = (float*)alloc((size_t)BLR * 128 * 4);
  float* Hinit = (float*)alloc((size_t)B_SZ * DI * NCH * DS * 4);

  float* dtbuf = xg;                        // [BLR, DI]
  float* ybuf  = xg + (size_t)BLR * DI;     // [BLR, DI]
  float* hloc  = x_inner;                   // [B,DI,NCH,DS]
  float* Ploc  = x_inner + (size_t)B_SZ * DI * NCH * DS;
  unsigned short* y_bf = xc_bf;

  const int ELT_G = 4096, ELT_B = 256;

  // 1) fp32 -> bf16 conversions (weights + input)
  k_cvt_bf16<<<ELT_G, ELT_B, 0, stream>>>(x, x_bf, (long)BLR * DM);
  k_cvt_bf16<<<ELT_G, ELT_B, 0, stream>>>(in_w, w_in_bf, (long)2 * DI * DM);
  k_cvt_bf16<<<ELT_G, ELT_B, 0, stream>>>(dt_w, dt_w_bf, (long)DI * DI);
  k_cvt_bf16<<<ELT_G, ELT_B, 0, stream>>>(B_w, bc_w_bf, (long)DS * DI);
  k_cvt_bf16<<<ELT_G, ELT_B, 0, stream>>>(C_w, bc_w_bf + (size_t)DS * DI, (long)DS * DI);
  k_cvt_bf16<<<ELT_G, ELT_B, 0, stream>>>(out_w, out_w_bf, (long)DM * DI);

  // 2) in_proj GEMM: [BLR,DM] x [2DI,DM]^T -> xg [BLR, 2DI]
  k_gemm_bf16<<<dim3((2 * DI) / BN, BLR / BM), 256, 0, stream>>>(x_bf, w_in_bf, xg, BLR, 2 * DI, DM);

  // 3) gating -> x_inner
  k_gate<<<ELT_G, ELT_B, 0, stream>>>(xg, x_inner, (long)BLR * DI);

  // 4) causal depthwise conv + RMSNorm -> xc_f32, xc_bf
  k_conv_rms<<<BLR, 256, 0, stream>>>(x_inner, conv_w, conv_b, norm_w, xc_f32, xc_bf);

  // 5) dt projection GEMM (overwrites xg low half) + softplus
  k_gemm_bf16<<<dim3(DI / BN, BLR / BM), 256, 0, stream>>>(xc_bf, dt_w_bf, dtbuf, BLR, DI, DI);
  k_softplus<<<ELT_G, ELT_B, 0, stream>>>(dtbuf, dt_b, (long)BLR * DI);

  // 6) fused B/C projection GEMM -> bct [BLR, 128] (B in 0..63, C in 64..127)
  k_gemm_bf16<<<dim3(128 / BN, BLR / BM), 256, 0, stream>>>(xc_bf, bc_w_bf, bct, BLR, 128, DI);

  // 7) chunked selective scan
  k_scan_local<<<dim3(DI, NCH, B_SZ), 64, 0, stream>>>(dtbuf, xc_f32, bct, A_log, hloc, Ploc);
  {
    const long tot = (long)B_SZ * DI * DS;
    k_scan_combine<<<(unsigned)((tot + 255) / 256), 256, 0, stream>>>(hloc, Ploc, Hinit);
  }
  k_scan_emit<<<dim3(DI, NCH, B_SZ), 64, 0, stream>>>(dtbuf, xc_f32, bct, A_log, Hinit, ybuf);

  // 8) skip connection + bf16 convert
  k_skip_cvt<<<ELT_G, ELT_B, 0, stream>>>(ybuf, xc_f32, Dp, y_bf, (long)BLR * DI);

  // 9) out_proj GEMM -> d_out [BLR, DM]
  k_gemm_bf16<<<dim3(DM / BN, BLR / BM), 256, 0, stream>>>(y_bf, out_w_bf, out, BLR, DM, DI);

  (void)in_sizes; (void)n_in; (void)out_size; (void)ws_size;
}
